// VectorField_35467839930473
// MI455X (gfx1250) — matrix operand order repaired and checked by reading, not compiled, not run on hardware
//
#include <hip/hip_runtime.h>

// MI455X / gfx1250 edge-MLP + LayerNorm.
// Memory-bound target (~550 MB @ 23.3 TB/s ~= 24 us); f16 WMMA 16x16x32 with
// f32 accumulate supplies >1.3 PFLOP/s so the tensor cores never gate HBM.

typedef __attribute__((ext_vector_type(16))) _Float16 v16h;
typedef __attribute__((ext_vector_type(8)))  _Float16 v8h;
typedef __attribute__((ext_vector_type(8)))  float    v8f;

#define N_EDGES     800000
#define NF          64
#define IN_DIM      224
#define LN_EPS      1e-5f

#define W1T_STRIDE  232   // halves; 464B row: 16B aligned, conflict-friendly
#define W2T_STRIDE  72    // halves; 144B row: 16B aligned
#define H_STRIDE    72    // halves; 144B row: 16B aligned
#define WAVES       8

__device__ __forceinline__ float silu_f(float x) {
  return x / (1.0f + __expf(-x));
}

// Layer-1 A fragment: 16x32 f16 tile, built from 2 aligned 32B f32 runs.
// lane<16 holds row=lane, K = {h*8..h*8+7, 16+h*8..16+h*8+7} with h=lane/16.
__device__ __forceinline__ v16h a_from_global(const float* __restrict__ p, int half) {
  const float4* q0 = (const float4*)(p + half * 8);
  const float4* q1 = (const float4*)(p + 16 + half * 8);
  float4 f0 = q0[0], f1 = q0[1];
  float4 f2 = q1[0], f3 = q1[1];
  v16h a;
  a[0]  = (_Float16)f0.x; a[1]  = (_Float16)f0.y; a[2]  = (_Float16)f0.z; a[3]  = (_Float16)f0.w;
  a[4]  = (_Float16)f1.x; a[5]  = (_Float16)f1.y; a[6]  = (_Float16)f1.z; a[7]  = (_Float16)f1.w;
  a[8]  = (_Float16)f2.x; a[9]  = (_Float16)f2.y; a[10] = (_Float16)f2.z; a[11] = (_Float16)f2.w;
  a[12] = (_Float16)f3.x; a[13] = (_Float16)f3.y; a[14] = (_Float16)f3.z; a[15] = (_Float16)f3.w;
  return a;
}

// B fragment: lane holds column n = lane%16 (+nt*16), K = {h*8..h*8+7, 16+h*8..16+h*8+7}.
__device__ __forceinline__ v16h b_from_lds(const _Float16* p) {
  v8h lo = *(const v8h*)(p);
  v8h hi = *(const v8h*)(p + 16);
  v16h b;
#pragma unroll
  for (int i = 0; i < 8; ++i) { b[i] = lo[i]; b[8 + i] = hi[i]; }
  return b;
}

// Layer-2 A fragment out of the per-wave f16 transpose tile.
__device__ __forceinline__ v16h a_from_lds(const _Float16* rowbase, int kc, int half) {
  v8h r1 = *(const v8h*)(rowbase + kc + half * 8);
  v8h r2 = *(const v8h*)(rowbase + kc + 16 + half * 8);
  v16h a;
#pragma unroll
  for (int i = 0; i < 8; ++i) { a[i] = r1[i]; a[8 + i] = r2[i]; }
  return a;
}

__global__ __launch_bounds__(256)
void VectorField_35467839930473_kernel(const float* __restrict__ node_scalars,
                                       const float* __restrict__ edge_feats,
                                       const float* __restrict__ rbf,
                                       const int*   __restrict__ src_idxs,
                                       const int*   __restrict__ dst_idxs,
                                       const float* __restrict__ W1,
                                       const float* __restrict__ b1,
                                       const float* __restrict__ W2,
                                       const float* __restrict__ b2,
                                       const float* __restrict__ ln_g,
                                       const float* __restrict__ ln_b,
                                       float* __restrict__ out) {
  __shared__ __align__(16) _Float16 w1t[64 * W1T_STRIDE];   // 29 KB
  __shared__ __align__(16) _Float16 w2t[64 * W2T_STRIDE];   // 9 KB
  __shared__ __align__(16) _Float16 hbuf[WAVES][16 * H_STRIDE]; // 18 KB

  const int tid = threadIdx.x;

  // ---- stage f16 transposed weights into LDS (shared by all 8 waves) ----
  for (int i = tid; i < IN_DIM * NF; i += 256) {
    int k = i >> 6, n = i & 63;               // W1 is [224][64] row-major
    w1t[n * W1T_STRIDE + k] = (_Float16)W1[i];
  }
  for (int i = tid; i < NF * NF; i += 256) {
    int k = i >> 6, n = i & 63;
    w2t[n * W2T_STRIDE + k] = (_Float16)W2[i];
  }
  __syncthreads();

  const int wave = tid >> 5;
  const int lane = tid & 31;
  const int half = lane >> 4;     // which 16-lane half
  const int lrow = lane & 15;
  const int e0   = (blockIdx.x * WAVES + wave) * 16;   // 16 edges per wave
  const int eA   = e0 + lrow;                          // A-matrix row edge

  const int s  = src_idxs[eA];
  const int dn = dst_idxs[eA];

  const float* pS  = node_scalars + (long)s  * 64;
  const float* pD  = node_scalars + (long)dn * 64;
  const float* pE  = edge_feats   + (long)eA * 64;
  const float* pR  = rbf          + (long)eA * 32;

  // ---- layer 1: [16 x 224] @ [224 x 64], K in 7 chunks of 32 ----
  v8f acc[4] = {};
  const float* bases[7] = { pS, pS + 32, pD, pD + 32, pE, pE + 32, pR };
#pragma unroll
  for (int kk = 0; kk < 7; ++kk) {
    v16h aF = a_from_global(bases[kk], half);
    const int kc = kk * 32;
#pragma unroll
    for (int nt = 0; nt < 4; ++nt) {
      v16h bF = b_from_lds(&w1t[(nt * 16 + lrow) * W1T_STRIDE + kc + half * 8]);
      acc[nt] = __builtin_amdgcn_wmma_f32_16x16x32_f16(
          false, aF, false, bF, (short)0, acc[nt], false, false);
    }
  }

  // ---- bias + SiLU, transpose through per-wave LDS tile as f16 ----
  _Float16* hme = &hbuf[wave][0];
#pragma unroll
  for (int nt = 0; nt < 4; ++nt) {
    const int   c  = nt * 16 + lrow;
    const float bb = b1[c];
#pragma unroll
    for (int j = 0; j < 8; ++j) {
      float x = acc[nt][j] + bb;                 // D layout: row = 8*half+j, col = c
      hme[(8 * half + j) * H_STRIDE + c] = (_Float16)silu_f(x);
    }
  }
  asm volatile("s_wait_dscnt 0" ::: "memory");   // wave-local store->load fence

  // ---- layer 2: [16 x 64] @ [64 x 64], K in 2 chunks of 32 ----
  v8f acc2[4] = {};
#pragma unroll
  for (int kk = 0; kk < 2; ++kk) {
    const int kc = kk * 32;
    v16h aF = a_from_lds(&hme[lrow * H_STRIDE], kc, half);
#pragma unroll
    for (int nt = 0; nt < 4; ++nt) {
      v16h bF = b_from_lds(&w2t[(nt * 16 + lrow) * W2T_STRIDE + kc + half * 8]);
      acc2[nt] = __builtin_amdgcn_wmma_f32_16x16x32_f16(
          false, aF, false, bF, (short)0, acc2[nt], false, false);
    }
  }

  // ---- bias + SiLU + residual, all in D layout ----
  float yv[4][8];
#pragma unroll
  for (int nt = 0; nt < 4; ++nt) {
    const int   c  = nt * 16 + lrow;
    const float bb = b2[c];
#pragma unroll
    for (int j = 0; j < 8; ++j) {
      float h  = silu_f(acc2[nt][j] + bb);
      float ef = edge_feats[(long)(e0 + 8 * half + j) * 64 + c];
      yv[nt][j] = ef + h;
    }
  }

  // ---- LayerNorm: row sums live across the 16 lanes of each half ----
  float s1[8], s2[8];
#pragma unroll
  for (int j = 0; j < 8; ++j) {
    float a0 = yv[0][j], a1 = yv[1][j], a2 = yv[2][j], a3 = yv[3][j];
    s1[j] = a0 + a1 + a2 + a3;
    s2[j] = a0 * a0 + a1 * a1 + a2 * a2 + a3 * a3;
  }
#pragma unroll
  for (int m = 1; m <= 8; m <<= 1) {
#pragma unroll
    for (int j = 0; j < 8; ++j) {
      s1[j] += __shfl_xor(s1[j], m, 32);
      s2[j] += __shfl_xor(s2[j], m, 32);
    }
  }
#pragma unroll
  for (int j = 0; j < 8; ++j) {
    float mu  = s1[j] * (1.0f / 64.0f);
    float var = s2[j] * (1.0f / 64.0f) - mu * mu;
    s1[j] = mu;
    s2[j] = rsqrtf(var + LN_EPS);
  }

  // ---- scale/shift + coalesced store (16 lanes -> 64B contiguous) ----
#pragma unroll
  for (int nt = 0; nt < 4; ++nt) {
    const int   c = nt * 16 + lrow;
    const float g = ln_g[c], bo = ln_b[c];
#pragma unroll
    for (int j = 0; j < 8; ++j) {
      out[(long)(e0 + 8 * half + j) * 64 + c] =
          (yv[nt][j] - s1[j]) * s2[j] * g + bo;
    }
  }
}

extern "C" void kernel_launch(void* const* d_in, const int* in_sizes, int n_in,
                              void* d_out, int out_size, void* d_ws, size_t ws_size,
                              hipStream_t stream) {
  (void)in_sizes; (void)n_in; (void)out_size; (void)d_ws; (void)ws_size;
  const float* node_scalars = (const float*)d_in[0];
  const float* edge_feats   = (const float*)d_in[1];
  const float* rbf          = (const float*)d_in[2];
  const int*   src_idxs     = (const int*)d_in[3];
  const int*   dst_idxs     = (const int*)d_in[4];
  const float* W1           = (const float*)d_in[5];
  const float* b1           = (const float*)d_in[6];
  const float* W2           = (const float*)d_in[7];
  const float* b2           = (const float*)d_in[8];
  const float* ln_g         = (const float*)d_in[9];
  const float* ln_b         = (const float*)d_in[10];
  float* out = (float*)d_out;

  // 800000 edges / (8 waves * 16 edges) = 6250 blocks exactly (no tail).
  VectorField_35467839930473_kernel<<<dim3(N_EDGES / (WAVES * 16)), dim3(256), 0, stream>>>(
      node_scalars, edge_feats, rbf, src_idxs, dst_idxs,
      W1, b1, W2, b2, ln_g, ln_b, out);
}
